// Attention_27874337751826
// MI455X (gfx1250) — matrix operand order repaired and checked by reading, not compiled, not run on hardware
//
#include <hip/hip_runtime.h>
#include <hip/hip_bf16.h>

// ---------------- CDNA5 WMMA types ----------------
typedef __attribute__((ext_vector_type(16))) __bf16 v16bf;
typedef __attribute__((ext_vector_type(8)))  __bf16 v8bf;
typedef __attribute__((ext_vector_type(8)))  float  v8f;

static constexpr int S_LEN  = 2048;
static constexpr int HID    = 2048;
static constexpr int NH     = 16;
static constexpr int NKV    = 4;
static constexpr int HD     = 128;
static constexpr int TOK    = 2 * S_LEN;        // 4096 tokens total
static constexpr int NQKV   = NH*HD + 2*NKV*HD; // 3072 fused QKV cols

__device__ __forceinline__ v16bf make_v16(v8bf lo, v8bf hi) {
  v16bf r;
#pragma unroll
  for (int i = 0; i < 8; ++i) { r[i] = lo[i]; r[i + 8] = hi[i]; }
  return r;
}

// Async DMA: global (16B) -> LDS, tracked by ASYNCcnt (no VGPR round-trip).
// Low 32 bits of a generic pointer to __shared__ are the LDS-relative address.
__device__ __forceinline__ void async_b128(const __bf16* gptr, __bf16* lptr) {
  unsigned lds = (unsigned)(size_t)lptr;
  unsigned long long ga = (unsigned long long)gptr;
  asm volatile("global_load_async_to_lds_b128 %0, %1, off"
               :: "v"(lds), "v"(ga) : "memory");
}
__device__ __forceinline__ void wait_async0() {
  asm volatile("s_wait_asynccnt 0" ::: "memory");
}

// ---------------- phase 0: precision / layout conversion ----------------
__global__ __launch_bounds__(256) void convert_x(const float* __restrict__ X,
                                                 __bf16* __restrict__ X16) {
  unsigned i = blockIdx.x * 256u + threadIdx.x;
  X16[i] = (__bf16)X[i];
}

// WqkvT[n][k], n in [0,3072): 0..2047 Wq cols, 2048..2559 Wk, 2560..3071 Wv
__global__ __launch_bounds__(256) void pack_wqkv(const float* __restrict__ Wq,
                                                 const float* __restrict__ Wk,
                                                 const float* __restrict__ Wv,
                                                 __bf16* __restrict__ Wt) {
  unsigned i = blockIdx.x * 256u + threadIdx.x; // 3072*2048 total
  unsigned n = i >> 11;
  unsigned k = i & 2047;
  float v;
  if (n < 2048u)      v = Wq[k * 2048u + n];
  else if (n < 2560u) v = Wk[k * 512u + (n - 2048u)];
  else                v = Wv[k * 512u + (n - 2560u)];
  Wt[i] = (__bf16)v;
}

__global__ __launch_bounds__(256) void pack_wo(const float* __restrict__ Wo,
                                               __bf16* __restrict__ WoT) {
  unsigned i = blockIdx.x * 256u + threadIdx.x; // 2048*2048
  unsigned n = i >> 11;
  unsigned k = i & 2047;
  WoT[i] = (__bf16)Wo[k * 2048u + n];
}

// ---------------- generic bf16 WMMA GEMM: C[M][N] = A[M][K] * Bt[N][K]^T ----
// 256 threads = 8 waves; WG tile 128x128; wave tile 32x64 (2x4 wmma tiles).
// Double-buffered LDS with async prefetch of tile i+1 under tile i's WMMAs.
static constexpr int LP = 40; // LDS pitch (bf16 elems), 80B = 16B aligned, conflict-friendly

template <int N, int K>
__global__ __launch_bounds__(256) void gemm_bf16(const __bf16* __restrict__ A,
                                                 const __bf16* __restrict__ Bt,
                                                 float* __restrict__ C) {
  __shared__ __align__(16) __bf16 As[2][128 * LP];
  __shared__ __align__(16) __bf16 Bs[2][128 * LP];

  const int tid  = threadIdx.x;
  const int lane = tid & 31;
  const int wave = tid >> 5;       // 0..7
  const int wm   = wave >> 1;      // 0..3  (M direction)
  const int wn   = wave & 1;       // 0..1  (N direction)
  const int half = (lane >> 4) & 1;
  const int l16  = lane & 15;

  const int m0 = blockIdx.y * 128;
  const int n0 = blockIdx.x * 128;

  v8f acc[2][4];
#pragma unroll
  for (int mt = 0; mt < 2; ++mt)
#pragma unroll
    for (int nt = 0; nt < 4; ++nt)
#pragma unroll
      for (int e = 0; e < 8; ++e) acc[mt][nt][e] = 0.f;

  // per-thread staging coordinates (2 chunks per tile per thread)
  const int c0r = (tid + 0)   >> 2, c0k = ((tid + 0)   & 3) * 8;
  const int c1r = (tid + 256) >> 2, c1k = ((tid + 256) & 3) * 8;

  auto issue = [&](int buf, int k0) {
    async_b128(&A [(unsigned)((m0 + c0r) * K + k0 + c0k)], &As[buf][c0r * LP + c0k]);
    async_b128(&A [(unsigned)((m0 + c1r) * K + k0 + c1k)], &As[buf][c1r * LP + c1k]);
    async_b128(&Bt[(unsigned)((n0 + c0r) * K + k0 + c0k)], &Bs[buf][c0r * LP + c0k]);
    async_b128(&Bt[(unsigned)((n0 + c1r) * K + k0 + c1k)], &Bs[buf][c1r * LP + c1k]);
  };

  constexpr int NIT = K / 32;
  issue(0, 0);

#pragma unroll 1
  for (int i = 0; i < NIT; ++i) {
    wait_async0();        // this wave's chunks of tile i have landed
    __syncthreads();      // all waves' chunks landed; prev-prev buffer reads done
    if (i + 1 < NIT) issue((i + 1) & 1, (i + 1) * 32);  // prefetch under compute

    const __bf16* as = As[i & 1];
    const __bf16* bs = Bs[i & 1];

    v16bf af[2], bfm[4];
#pragma unroll
    for (int mt = 0; mt < 2; ++mt) {
      int row = wm * 32 + mt * 16 + l16;     // A: m = lane%16
      v8bf lo = *(const v8bf*)&as[row * LP + half * 8];
      v8bf hi = *(const v8bf*)&as[row * LP + 16 + half * 8];
      af[mt] = make_v16(lo, hi);
    }
#pragma unroll
    for (int nt = 0; nt < 4; ++nt) {
      int row = wn * 64 + nt * 16 + l16;     // B: n = lane%16, K = half*8 + 0..7, 16 + half*8 + 0..7
      v8bf lo = *(const v8bf*)&bs[row * LP + half * 8];
      v8bf hi = *(const v8bf*)&bs[row * LP + 16 + half * 8];
      bfm[nt] = make_v16(lo, hi);
    }
#pragma unroll
    for (int mt = 0; mt < 2; ++mt)
#pragma unroll
      for (int nt = 0; nt < 4; ++nt)
        acc[mt][nt] = __builtin_amdgcn_wmma_f32_16x16x32_bf16(
            false, af[mt], false, bfm[nt], (short)0, acc[mt][nt], false, false);
  }

  // C layout: VGPR r -> row r + 8*half, col = lane%16.
  // Single base + compile-time offsets -> immediate-offset saddr stores.
  const unsigned cbase = (unsigned)((m0 + wm * 32 + half * 8) * N + n0 + wn * 64 + l16);
#pragma unroll
  for (int mt = 0; mt < 2; ++mt)
#pragma unroll
    for (int nt = 0; nt < 4; ++nt)
#pragma unroll
      for (int r = 0; r < 8; ++r)
        C[cbase + (unsigned)((mt * 16 + r) * N + nt * 16)] = acc[mt][nt][r];
}

// ---------------- phase 1b: RMSNorm + RoPE epilogue ----------------
// grid (TOK, 24): heads 0..15 -> Q, 16..19 -> K, 20..23 -> V
__global__ __launch_bounds__(128) void norm_rope(const float* __restrict__ QKV,
                                                 const float* __restrict__ qw,
                                                 const float* __restrict__ kw,
                                                 __bf16* __restrict__ Qr,   // [B][NH][S][D]
                                                 __bf16* __restrict__ Kr,   // [B][NKV][S][D]
                                                 __bf16* __restrict__ Vt) { // [B][NKV][D][S]
  __shared__ float sh[128];
  __shared__ float red[128];
  const int t = blockIdx.x, head = blockIdx.y, d = threadIdx.x;
  const int b = t >> 11, s = t & 2047;
  float v = QKV[(unsigned)(t * NQKV + head * 128 + d)];

  if (head >= 20) { // V: no norm/rope; store transposed for PV B-fragments
    int hk = head - 20;
    Vt[(unsigned)(((b * NKV + hk) * HD + d) * S_LEN + s)] = (__bf16)v;
    return;
  }
  // RMSNorm over 128
  red[d] = v * v;
  __syncthreads();
#pragma unroll
  for (int o = 64; o > 0; o >>= 1) {
    if (d < o) red[d] += red[d + o];
    __syncthreads();
  }
  float rr = rsqrtf(red[0] * (1.f / 128.f) + 1e-6f);
  float w  = (head < 16) ? qw[d] : kw[d];
  float y  = v * rr * w;
  sh[d] = y;
  __syncthreads();
  // RoPE (rotate_half): pair (d, d+64), freq index = d mod 64
  int j = d & 63;
  float f = (float)s * __powf(1000000.0f, -(float)j * (1.0f / 64.0f));
  float sn, cs;
  __sincosf(f, &sn, &cs);
  float partner = (d < 64) ? sh[d + 64] : sh[d - 64];
  float out = (d < 64) ? (y * cs - partner * sn) : (y * cs + partner * sn);
  if (head < 16)
    Qr[(unsigned)(((b * NH + head) * S_LEN + s) * HD + d)] = (__bf16)out;
  else
    Kr[(unsigned)(((b * NKV + (head - 16)) * S_LEN + s) * HD + d)] = (__bf16)out;
}

// ---------------- phase 2: flash attention with soft-cap ----------------
// grid (S/64, NH, B); 128 threads = 4 waves; wave owns 16 q rows.
// K/V tiles double-buffered; block kb+1 DMA'd under block kb's compute.
static constexpr int QP = 136; // 128+8 pitch
static constexpr int VP = 72;  // 64+8 pitch

__global__ __launch_bounds__(128) void flash_attn(const __bf16* __restrict__ Qr,
                                                  const __bf16* __restrict__ Kr,
                                                  const __bf16* __restrict__ Vt,
                                                  __bf16* __restrict__ Out) { // [TOK][NH*HD]
  __shared__ __align__(16) __bf16 Qs [64 * QP];
  __shared__ __align__(16) __bf16 Ks [2][64 * QP];
  __shared__ __align__(16) __bf16 Vts[2][128 * VP];
  __shared__ __align__(16) __bf16 Ps [64 * VP];

  const int tid  = threadIdx.x;
  const int lane = tid & 31;
  const int wave = tid >> 5;        // 0..3
  const int half = (lane >> 4) & 1;
  const int l16  = lane & 15;

  const int qt = blockIdx.x, h = blockIdx.y, b = blockIdx.z;
  const int hk = h >> 2;            // GQA: repeat_interleave(4)
  const int q0 = qt * 64;

  const __bf16* Qbase = Qr + (size_t)(b * NH + h)   * S_LEN * HD;
  const __bf16* Kbase = Kr + (size_t)(b * NKV + hk) * S_LEN * HD;
  const __bf16* Vbase = Vt + (size_t)(b * NKV + hk) * HD * S_LEN;

  auto issue_kv = [&](int buf, int kb) {
#pragma unroll
    for (int i = 0; i < 8; ++i) {
      int c = tid + i * 128;
      int r = c >> 4, kc = (c & 15) * 8;
      async_b128(&Kbase[(unsigned)((kb * 64 + r) * HD + kc)], &Ks[buf][r * QP + kc]);
    }
#pragma unroll
    for (int i = 0; i < 8; ++i) {
      int c = tid + i * 128;
      int d = c >> 3, kc = (c & 7) * 8;
      async_b128(&Vbase[(unsigned)(d * S_LEN + kb * 64 + kc)], &Vts[buf][d * VP + kc]);
    }
  };

  // stage Q tile (64x128) and first K/V block together
#pragma unroll
  for (int i = 0; i < 8; ++i) {
    int c = tid + i * 128;
    int r = c >> 4, kc = (c & 15) * 8;
    async_b128(&Qbase[(unsigned)((q0 + r) * HD + kc)], &Qs[r * QP + kc]);
  }
  issue_kv(0, 0);
  wait_async0();
  __syncthreads();

  // Q A-fragments kept live across key loop (4 k-steps over D=128)
  v16bf qf[4];
  const int qrow = wave * 16 + l16;
#pragma unroll
  for (int ks = 0; ks < 4; ++ks) {
    v8bf lo = *(const v8bf*)&Qs[qrow * QP + ks * 32 + half * 8];
    v8bf hi = *(const v8bf*)&Qs[qrow * QP + ks * 32 + 16 + half * 8];
    qf[ks] = make_v16(lo, hi);
  }

  v8f o[8];
  float mrow[8], lrow[8];
#pragma unroll
  for (int dn = 0; dn < 8; ++dn)
#pragma unroll
    for (int e = 0; e < 8; ++e) o[dn][e] = 0.f;
#pragma unroll
  for (int r = 0; r < 8; ++r) { mrow[r] = -1e30f; lrow[r] = 0.f; }

  const float scale = 0.08838834764831845f; // 128^-0.5
  const int nkb = qt + 1;                    // causal: key blocks 0..qt

#pragma unroll 1
  for (int kb = 0; kb < nkb; ++kb) {
    if (kb) {
      wait_async0();      // this wave's block-kb chunks landed
      __syncthreads();    // all waves landed; buffer (kb+1)&1 reads finished
    }
    if (kb + 1 < nkb) issue_kv((kb + 1) & 1, kb + 1);  // prefetch under compute

    const __bf16* ks_ = Ks[kb & 1];
    const __bf16* vs_ = Vts[kb & 1];

    // scores: S = Q K^T  (4 key n-tiles x 4 d k-steps)
    v8f sc[4];
#pragma unroll
    for (int nt = 0; nt < 4; ++nt) {
#pragma unroll
      for (int e = 0; e < 8; ++e) sc[nt][e] = 0.f;
#pragma unroll
      for (int ks = 0; ks < 4; ++ks) {
        int krow = nt * 16 + l16;    // key index = B n
        v8bf lo = *(const v8bf*)&ks_[krow * QP + ks * 32 + half * 8];
        v8bf hi = *(const v8bf*)&ks_[krow * QP + ks * 32 + 16 + half * 8];
        sc[nt] = __builtin_amdgcn_wmma_f32_16x16x32_bf16(
            false, qf[ks], false, make_v16(lo, hi), (short)0, sc[nt], false, false);
      }
    }

    // soft-cap + causal mask + online softmax (row = 8*half + rv, cols across l16)
#pragma unroll
    for (int rv = 0; rv < 8; ++rv) {
      int qg = q0 + wave * 16 + half * 8 + rv;
      float mx = mrow[rv];
#pragma unroll
      for (int nt = 0; nt < 4; ++nt) {
        int kg = kb * 64 + nt * 16 + l16;
        float s = sc[nt][rv] * scale;
        s = 50.0f * tanhf(s * 0.02f);
        s = (kg > qg) ? -1e30f : s;
        sc[nt][rv] = s;
        mx = fmaxf(mx, s);
      }
#pragma unroll
      for (int off = 8; off > 0; off >>= 1) mx = fmaxf(mx, __shfl_xor(mx, off, 32));
      float alpha = __expf(mrow[rv] - mx);
      mrow[rv] = mx;
      float rsum = 0.f;
#pragma unroll
      for (int nt = 0; nt < 4; ++nt) {
        float p = __expf(sc[nt][rv] - mx);
        sc[nt][rv] = p;
        rsum += p;
      }
#pragma unroll
      for (int off = 8; off > 0; off >>= 1) rsum += __shfl_xor(rsum, off, 32);
      lrow[rv] = lrow[rv] * alpha + rsum;
#pragma unroll
      for (int dn = 0; dn < 8; ++dn) o[dn][rv] *= alpha;
    }

    // C-layout -> A-layout reshape of P via per-wave LDS region
#pragma unroll
    for (int nt = 0; nt < 4; ++nt)
#pragma unroll
      for (int rv = 0; rv < 8; ++rv)
        Ps[(wave * 16 + half * 8 + rv) * VP + nt * 16 + l16] = (__bf16)sc[nt][rv];

    // O += P V   (2 key k-steps x 8 d n-tiles)
#pragma unroll
    for (int ks2 = 0; ks2 < 2; ++ks2) {
      v8bf plo = *(const v8bf*)&Ps[(wave * 16 + l16) * VP + ks2 * 32 + half * 8];
      v8bf phi = *(const v8bf*)&Ps[(wave * 16 + l16) * VP + ks2 * 32 + 16 + half * 8];
      v16bf pf = make_v16(plo, phi);
#pragma unroll
      for (int dn = 0; dn < 8; ++dn) {
        v8bf blo = *(const v8bf*)&vs_[(dn * 16 + l16) * VP + ks2 * 32 + half * 8];
        v8bf bhi = *(const v8bf*)&vs_[(dn * 16 + l16) * VP + ks2 * 32 + 16 + half * 8];
        o[dn] = __builtin_amdgcn_wmma_f32_16x16x32_bf16(
            false, pf, false, make_v16(blo, bhi), (short)0, o[dn], false, false);
      }
    }
  }

  // normalize + store bf16, row-major [token][h*128+d] (A operand of out-proj).
  // Single base + compile-time offsets -> immediate-offset stores.
  const unsigned obase =
      (unsigned)((b * S_LEN + q0 + wave * 16 + half * 8) * (NH * HD) + h * HD + l16);
#pragma unroll
  for (int dn = 0; dn < 8; ++dn)
#pragma unroll
    for (int rv = 0; rv < 8; ++rv)
      Out[obase + (unsigned)(rv * (NH * HD) + dn * 16)] = (__bf16)(o[dn][rv] / lrow[rv]);
}

// ---------------- launcher ----------------
extern "C" void kernel_launch(void* const* d_in, const int* in_sizes, int n_in,
                              void* d_out, int out_size, void* d_ws, size_t ws_size,
                              hipStream_t stream) {
  const float* hs = (const float*)d_in[0];
  const float* Wq = (const float*)d_in[1];
  const float* Wk = (const float*)d_in[2];
  const float* Wv = (const float*)d_in[3];
  const float* Wo = (const float*)d_in[4];
  const float* qw = (const float*)d_in[5];
  const float* kw = (const float*)d_in[6];
  float* out = (float*)d_out;

  char* base = (char*)d_ws;
  size_t off = 0;
  auto take = [&](size_t n) -> char* {
    char* p = base + off;
    off += (n + 255) & ~(size_t)255;
    return p;
  };
  __bf16* X16   = (__bf16*)take((size_t)TOK * HID * 2);        // 16 MB
  __bf16* WqkvT = (__bf16*)take((size_t)NQKV * HID * 2);       // 12 MB
  __bf16* WoT   = (__bf16*)take((size_t)HID * HID * 2);        // 8 MB
  float*  QKV   = (float*) take((size_t)TOK * NQKV * 4);       // 48 MB
  __bf16* Qr    = (__bf16*)take((size_t)2 * NH  * S_LEN * HD * 2);
  __bf16* Kr    = (__bf16*)take((size_t)2 * NKV * S_LEN * HD * 2);
  __bf16* Vt    = (__bf16*)take((size_t)2 * NKV * HD * S_LEN * 2);
  __bf16* AO    = (__bf16*)take((size_t)TOK * NH * HD * 2);    // 16 MB

  convert_x<<<(TOK * HID) / 256, 256, 0, stream>>>(hs, X16);
  pack_wqkv<<<(NQKV * HID) / 256, 256, 0, stream>>>(Wq, Wk, Wv, WqkvT);
  pack_wo<<<(HID * HID) / 256, 256, 0, stream>>>(Wo, WoT);

  gemm_bf16<NQKV, HID><<<dim3(NQKV / 128, TOK / 128), 256, 0, stream>>>(X16, WqkvT, QKV);

  norm_rope<<<dim3(TOK, 24), 128, 0, stream>>>(QKV, qw, kw, Qr, Kr, Vt);

  flash_attn<<<dim3(S_LEN / 64, NH, 2), 128, 0, stream>>>(Qr, Kr, Vt, AO);

  gemm_bf16<HID, NH * HD><<<dim3(HID / 128, TOK / 128), 256, 0, stream>>>(AO, WoT, out);
}
